// FeedForward_49563922595878
// MI455X (gfx1250) — matrix operand order repaired and checked by reading, not compiled, not run on hardware
//
#include <hip/hip_runtime.h>
#include <hip/hip_bf16.h>

typedef __bf16 v16bf __attribute__((ext_vector_type(16)));
typedef float  v8f   __attribute__((ext_vector_type(8)));

#define T_TOK 512
#define DIM_D 1024
#define DIM_F 4096
#define N_EXP 8
#define RANK  16
#define LORA_SCALE 2.0f
#define LDT 48   // padded LDS row (bf16): 96B per row -> 16B aligned, staggered banks

// ---------------------------------------------------------------------------
// Fragment loaders per CDNA5 WMMA VGPR layouts (CDNA5 ISA), 16B LDS accesses.
// A (16x32 bf16, MxK): lane m: K 0..7 & 16..23 ; lane m+16: K 8..15 & 24..31
// ---------------------------------------------------------------------------
__device__ inline v16bf load_a_frag(const __bf16* __restrict__ base, int lane) {
    const int m = lane & 15;
    const int h = lane >> 4;
    const __bf16* row = base + m * LDT;
    union { uint4 u[2]; v16bf v; } r;
    r.u[0] = *(const uint4*)(row + h * 8);
    r.u[1] = *(const uint4*)(row + 16 + h * 8);
    return r.v;
}

// B (32x16 bf16, KxN) from W stored [N][K]: lane n: K 0..7 & 16..23 ; lane n+16: K 8..15 & 24..31
__device__ inline v16bf load_b_frag(const __bf16* __restrict__ base, int lane) {
    const int n = lane & 15;
    const int h = lane >> 4;
    const __bf16* row = base + n * LDT + h * 8;
    union { uint4 u[2]; v16bf v; } r;
    r.u[0] = *(const uint4*)(row);
    r.u[1] = *(const uint4*)(row + 16);
    return r.v;
}

// ---------------------------------------------------------------------------
// K0: RMSNorm + router (softmax -> top2 -> renormalize)
// ---------------------------------------------------------------------------
__global__ __launch_bounds__(256) void norm_router_kernel(
    const float* __restrict__ x, const float* __restrict__ norm_w,
    const float* __restrict__ gate_w,
    __bf16* __restrict__ t_bf, int* __restrict__ tok_i, float* __restrict__ tok_w)
{
    __shared__ float nx[DIM_D];
    __shared__ float red[8];
    __shared__ float logits[N_EXP];

    const int t = blockIdx.x;
    const int tid = threadIdx.x;
    const int lane = tid & 31;
    const int wave = tid >> 5;

    float4 xv = *(const float4*)(x + t * DIM_D + tid * 4);
    float ss = xv.x * xv.x + xv.y * xv.y + xv.z * xv.z + xv.w * xv.w;
#pragma unroll
    for (int m = 16; m >= 1; m >>= 1) ss += __shfl_xor(ss, m, 32);
    if (lane == 0) red[wave] = ss;
    __syncthreads();
    if (tid == 0) {
        float s = 0.f;
#pragma unroll
        for (int i = 0; i < 8; ++i) s += red[i];
        red[0] = rsqrtf(s / (float)DIM_D + 1e-6f);
    }
    __syncthreads();
    const float inv = red[0];

    const int d0 = tid * 4;
    float4 nv;
    nv.x = xv.x * inv * norm_w[d0 + 0];
    nv.y = xv.y * inv * norm_w[d0 + 1];
    nv.z = xv.z * inv * norm_w[d0 + 2];
    nv.w = xv.w * inv * norm_w[d0 + 3];
    nx[d0 + 0] = nv.x; nx[d0 + 1] = nv.y; nx[d0 + 2] = nv.z; nx[d0 + 3] = nv.w;
    __bf16* trow = t_bf + (size_t)t * DIM_D + d0;
    trow[0] = (__bf16)nv.x; trow[1] = (__bf16)nv.y;
    trow[2] = (__bf16)nv.z; trow[3] = (__bf16)nv.w;
    __syncthreads();

    {
        const int e = wave;
        const float* gw = gate_w + (size_t)e * DIM_D;
        float p = 0.f;
        for (int d = lane; d < DIM_D; d += 32) p += nx[d] * gw[d];
#pragma unroll
        for (int m = 16; m >= 1; m >>= 1) p += __shfl_xor(p, m, 32);
        if (lane == 0) logits[e] = p;
    }
    __syncthreads();

    if (tid == 0) {
        float mx = logits[0];
#pragma unroll
        for (int e = 1; e < N_EXP; ++e) mx = fmaxf(mx, logits[e]);
        float pe[N_EXP]; float sum = 0.f;
#pragma unroll
        for (int e = 0; e < N_EXP; ++e) { pe[e] = __expf(logits[e] - mx); sum += pe[e]; }
#pragma unroll
        for (int e = 0; e < N_EXP; ++e) pe[e] /= sum;
        int i0 = 0;
#pragma unroll
        for (int e = 1; e < N_EXP; ++e) if (pe[e] > pe[i0]) i0 = e;
        int i1 = (i0 == 0) ? 1 : 0;
#pragma unroll
        for (int e = 0; e < N_EXP; ++e) if (e != i0 && pe[e] > pe[i1]) i1 = e;
        const float sw = pe[i0] + pe[i1];
        tok_i[t * 2 + 0] = i0; tok_i[t * 2 + 1] = i1;
        tok_w[t * 2 + 0] = pe[i0] / sw; tok_w[t * 2 + 1] = pe[i1] / sw;
    }
}

// ---------------------------------------------------------------------------
// K1: counting sort of the 1024 (token,slot) pairs by expert. Single block.
// perm[] entries encode idx = t*2 + s.
// ---------------------------------------------------------------------------
__global__ __launch_bounds__(256) void sort_kernel(
    const int* __restrict__ tok_i, int* __restrict__ eoff,
    int* __restrict__ ecnt, int* __restrict__ perm)
{
    __shared__ int cnt[N_EXP];
    __shared__ int off[N_EXP];
    const int tid = threadIdx.x;
    if (tid < N_EXP) cnt[tid] = 0;
    __syncthreads();
    for (int idx = tid; idx < T_TOK * 2; idx += 256) atomicAdd(&cnt[tok_i[idx]], 1);
    __syncthreads();
    if (tid == 0) {
        int a = 0;
#pragma unroll
        for (int e = 0; e < N_EXP; ++e) { off[e] = a; a += cnt[e]; }
    }
    __syncthreads();
    if (tid < N_EXP) { eoff[tid] = off[tid]; ecnt[tid] = cnt[tid]; }
    __syncthreads();
    for (int idx = tid; idx < T_TOK * 2; idx += 256) {
        int e = tok_i[idx];
        int p = atomicAdd(&off[e], 1);
        perm[p] = idx;
    }
}

// ---------------------------------------------------------------------------
// K2: u[e][t][r] = sum_d t[t][d] * a[e][r][d]   (WMMA, M=512 N=16 K=1024)
// ---------------------------------------------------------------------------
__global__ __launch_bounds__(32) void lora_u_kernel(
    const __bf16* __restrict__ t_bf,
    const float* __restrict__ a1, const float* __restrict__ a3,
    float* __restrict__ u1, float* __restrict__ u3)
{
    const int lane = threadIdx.x & 31;
    const int m0 = blockIdx.x * 16;
    const int e  = blockIdx.y;
    const float* A = blockIdx.z ? a3 : a1;
    float* U = blockIdx.z ? u3 : u1;

    const int am = lane & 15, ah = lane >> 4;
    v8f acc = {};
    for (int k0 = 0; k0 < DIM_D; k0 += 32) {
        const __bf16* arow = t_bf + (size_t)(m0 + am) * DIM_D + k0;
        union { uint4 u[2]; v16bf v; } af;
        af.u[0] = *(const uint4*)(arow + ah * 8);
        af.u[1] = *(const uint4*)(arow + 16 + ah * 8);
        const float* wrow = A + ((size_t)e * RANK + am) * DIM_D + k0 + ah * 8;
        v16bf bfv;
#pragma unroll
        for (int i = 0; i < 16; ++i) bfv[i] = (__bf16)wrow[i < 8 ? i : i + 8];
        acc = __builtin_amdgcn_wmma_f32_16x16x32_bf16(false, af.v, false, bfv,
                                                      (short)0, acc, false, false);
    }
    const int cn = lane & 15, ch = lane >> 4;
#pragma unroll
    for (int v = 0; v < 8; ++v) {
        int m = m0 + v + 8 * ch;
        U[((size_t)e * T_TOK + m) * RANK + cn] = acc[v];
    }
}

// ---------------------------------------------------------------------------
// K3: C[M,N] = A[M,K](bf16) * W[N,K](f32 -> bf16)^T, f32 out.
// Block tile 64 x BN, 8 waves, wave tile (MI*16) x (NI*16).
// Double-buffered LDS ping-pong, 16B staging, prefetch on streamed W.
// ---------------------------------------------------------------------------
template <int BN, int MI, int NI>
__global__ __launch_bounds__(256) void wmma_gemm_abt(
    const __bf16* __restrict__ A, const float* __restrict__ W,
    float* __restrict__ C, int M, int N, int K)
{
    constexpr int BM = 64;
    constexpr int WCOLS = BN / (NI * 16);
    constexpr int WROWS = 8 / WCOLS;
    static_assert(WROWS * MI * 16 == BM, "bad tile");
    static_assert(WROWS * WCOLS == 8, "bad wave grid");

    __shared__ __bf16 As[2][BM][LDT];
    __shared__ __bf16 Ws[2][BN][LDT];

    const int tid  = threadIdx.x;
    const int lane = tid & 31;
    const int wave = tid >> 5;
    const int wy = wave / WCOLS;
    const int wx = wave % WCOLS;
    const int m0 = blockIdx.y * BM;
    const int n0 = blockIdx.x * BN;

    const int ar = tid >> 2;         // 0..63
    const int ac = (tid & 3) * 8;    // 0,8,16,24

    v8f acc[MI][NI] = {};

    const int nsteps = K / 32;

    // staging helper (as lambda): stage k-tile `ks` into buffer `buf`
    auto stage = [&](int ks, int buf) {
        const int k0 = ks * 32;
        {   // A tile: 8 bf16 per thread, 16B vector
            const __bf16* src = A + (size_t)(m0 + ar) * K + k0 + ac;
            *(uint4*)&As[buf][ar][ac] = *(const uint4*)src;
        }
        // W tile: BN/128 chunks of 16 f32 per thread
#pragma unroll
        for (int i = 0; i < BN / 128; ++i) {
            const int c = tid + i * 256;
            const int row = c >> 1;
            const int col = (c & 1) * 16;
            const float* src = W + (size_t)(n0 + row) * K + k0 + col;
            alignas(16) __bf16 tmp[16];
#pragma unroll
            for (int q = 0; q < 16; q += 4) {
                float4 f = *(const float4*)(src + q);
                tmp[q + 0] = (__bf16)f.x; tmp[q + 1] = (__bf16)f.y;
                tmp[q + 2] = (__bf16)f.z; tmp[q + 3] = (__bf16)f.w;
            }
            *(uint4*)&Ws[buf][row][col]     = *(const uint4*)&tmp[0];
            *(uint4*)&Ws[buf][row][col + 8] = *(const uint4*)&tmp[8];
        }
    };

    stage(0, 0);
    for (int s = 0; s < nsteps; ++s) {
        const int p = s & 1;
        __syncthreads();
        if (s + 2 < nsteps) {
            // prefetch streamed weights two steps ahead (global_prefetch_b8)
            __builtin_prefetch(W + (size_t)(n0 + (tid >> 1)) * K + (s + 2) * 32, 0, 1);
        }
        if (s + 1 < nsteps) stage(s + 1, p ^ 1);

        v16bf af[MI], bfv[NI];
#pragma unroll
        for (int mi = 0; mi < MI; ++mi)
            af[mi] = load_a_frag(&As[p][wy * MI * 16 + mi * 16][0], lane);
#pragma unroll
        for (int ni = 0; ni < NI; ++ni)
            bfv[ni] = load_b_frag(&Ws[p][wx * NI * 16 + ni * 16][0], lane);
#pragma unroll
        for (int mi = 0; mi < MI; ++mi)
#pragma unroll
            for (int ni = 0; ni < NI; ++ni)
                acc[mi][ni] = __builtin_amdgcn_wmma_f32_16x16x32_bf16(
                    false, af[mi], false, bfv[ni], (short)0, acc[mi][ni], false, false);
    }

    const int cn = lane & 15, ch = lane >> 4;
#pragma unroll
    for (int mi = 0; mi < MI; ++mi)
#pragma unroll
        for (int ni = 0; ni < NI; ++ni)
#pragma unroll
            for (int v = 0; v < 8; ++v) {
                int m = m0 + wy * MI * 16 + mi * 16 + v + 8 * ch;
                int n = n0 + wx * NI * 16 + ni * 16 + cn;
                C[(size_t)m * N + n] = acc[mi][ni][v];
            }
}

// ---------------------------------------------------------------------------
// K4: act: grouped by expert; b1/b3 rank-rows staged once per block.
// grid (E, F/64). 256 threads = 4 tokens x 64 f.
// ---------------------------------------------------------------------------
__global__ __launch_bounds__(256) void act_sorted_kernel(
    const float* __restrict__ base1, const float* __restrict__ base3,
    const float* __restrict__ u1, const float* __restrict__ u3,
    const float* __restrict__ b1, const float* __restrict__ b3,
    const int* __restrict__ eoff, const int* __restrict__ ecnt,
    const int* __restrict__ perm, __bf16* __restrict__ act)
{
    __shared__ float B1s[64][RANK];
    __shared__ float B3s[64][RANK];
    __shared__ float Us[4][2][RANK];

    const int e  = blockIdx.x;
    const int f0 = blockIdx.y * 64;
    const int tid = threadIdx.x;

    for (int i = tid; i < 64 * RANK; i += 256) {
        const int fl = i >> 4, r = i & 15;
        B1s[fl][r] = b1[((size_t)e * DIM_F + f0 + fl) * RANK + r];
        B3s[fl][r] = b3[((size_t)e * DIM_F + f0 + fl) * RANK + r];
    }

    const int n = ecnt[e], base = eoff[e];
    const int tk = tid >> 6;      // token-in-batch 0..3
    const int fl = tid & 63;

    for (int b0 = 0; b0 < n; b0 += 4) {
        __syncthreads();  // also covers B1s/B3s on first iteration
        if (tid < 128) {
            const int ti = (tid >> 5) & 3;
            const int br = (tid >> 4) & 1;
            const int r  = tid & 15;
            if (b0 + ti < n) {
                const int idx = perm[base + b0 + ti];
                const int t = idx >> 1;
                const float* up = (br ? u3 : u1) + ((size_t)e * T_TOK + t) * RANK;
                Us[ti][br][r] = up[r];
            }
        }
        __syncthreads();
        if (b0 + tk < n) {
            const int idx = perm[base + b0 + tk];
            const int t = idx >> 1, s = idx & 1;
            const int f = f0 + fl;
            float d1 = 0.f, d3 = 0.f;
#pragma unroll
            for (int r = 0; r < RANK; ++r) {
                d1 += Us[tk][0][r] * B1s[fl][r];
                d3 += Us[tk][1][r] * B3s[fl][r];
            }
            const float h1 = base1[(size_t)t * DIM_F + f] + LORA_SCALE * d1;
            const float h3 = base3[(size_t)t * DIM_F + f] + LORA_SCALE * d3;
            const float sg = 1.0f / (1.0f + __expf(-h1));
            act[((size_t)s * T_TOK + t) * DIM_F + f] = (__bf16)(h1 * sg * h3);
        }
    }
}

// ---------------------------------------------------------------------------
// K5: zero the v accumulator (atomic target) each launch
// ---------------------------------------------------------------------------
__global__ __launch_bounds__(256) void zero_kernel(float* __restrict__ p, int n) {
    const int i = blockIdx.x * 256 + threadIdx.x;
    if (i < n) p[i] = 0.f;
}

// ---------------------------------------------------------------------------
// K6: v[s][t][r] += sum_{f in chunk} act[s][t][f] * a2[e][r][f]
// grid (E, F/256). 256 threads = 16 r x 16 lanes; a2 chunk staged in LDS.
// ---------------------------------------------------------------------------
__global__ __launch_bounds__(256) void v_sorted_kernel(
    const __bf16* __restrict__ act, const float* __restrict__ a2,
    const int* __restrict__ eoff, const int* __restrict__ ecnt,
    const int* __restrict__ perm, float* __restrict__ v)
{
    __shared__ float A2s[RANK][256];

    const int e  = blockIdx.x;
    const int f0 = blockIdx.y * 256;
    const int tid = threadIdx.x;

    for (int i = tid; i < RANK * 256; i += 256) {
        const int r = i >> 8, f = i & 255;
        A2s[r][f] = a2[((size_t)e * RANK + r) * DIM_F + f0 + f];
    }
    __syncthreads();

    const int n = ecnt[e], base = eoff[e];
    const int r = tid >> 4, j = tid & 15;

    for (int ti = 0; ti < n; ++ti) {
        const int idx = perm[base + ti];
        const int t = idx >> 1, s = idx & 1;
        const __bf16* arow = act + ((size_t)s * T_TOK + t) * DIM_F + f0;
        float p = 0.f;
#pragma unroll
        for (int f = j; f < 256; f += 16) p += (float)arow[f] * A2s[r][f];
#pragma unroll
        for (int m = 1; m < 16; m <<= 1) p += __shfl_xor(p, m, 32);
        if (j == 0) atomicAdd(&v[((size_t)s * T_TOK + t) * RANK + r], p);
    }
}

// ---------------------------------------------------------------------------
// K7: out[t][d] = sum_s w_s * (bout[s][t][d] + 2 * sum_r v[s][t][r]*b2[e][d][r])
// ---------------------------------------------------------------------------
__global__ __launch_bounds__(256) void combine_kernel(
    const float* __restrict__ bout, const float* __restrict__ v,
    const float* __restrict__ b2, const int* __restrict__ tok_i,
    const float* __restrict__ tok_w, float* __restrict__ out)
{
    const int t = blockIdx.x, tid = threadIdx.x;
    float accv[4] = {0.f, 0.f, 0.f, 0.f};
#pragma unroll
    for (int s = 0; s < 2; ++s) {
        const int e = tok_i[t * 2 + s];
        const float wgt = tok_w[t * 2 + s];
        float V[RANK];
        const float* vp = v + ((size_t)s * T_TOK + t) * RANK;
#pragma unroll
        for (int r = 0; r < RANK; ++r) V[r] = vp[r];
        const float* brow = bout + ((size_t)s * T_TOK + t) * DIM_D;
#pragma unroll
        for (int i = 0; i < 4; ++i) {
            const int d = tid + i * 256;
            const float* b2r = b2 + ((size_t)e * DIM_D + d) * RANK;
            float dl = 0.f;
#pragma unroll
            for (int r = 0; r < RANK; ++r) dl += V[r] * b2r[r];
            accv[i] += wgt * (brow[d] + LORA_SCALE * dl);
        }
    }
#pragma unroll
    for (int i = 0; i < 4; ++i) out[(size_t)t * DIM_D + tid + i * 256] = accv[i];
}

// ---------------------------------------------------------------------------
extern "C" void kernel_launch(void* const* d_in, const int* in_sizes, int n_in,
                              void* d_out, int out_size, void* d_ws, size_t ws_size,
                              hipStream_t stream) {
    const float* x      = (const float*)d_in[0];
    const float* norm_w = (const float*)d_in[1];
    const float* w1     = (const float*)d_in[2];
    const float* w3     = (const float*)d_in[3];
    const float* w2     = (const float*)d_in[4];
    const float* gate_w = (const float*)d_in[5];
    const float* a1     = (const float*)d_in[6];
    const float* b1     = (const float*)d_in[7];
    const float* a3     = (const float*)d_in[8];
    const float* b3     = (const float*)d_in[9];
    const float* a2     = (const float*)d_in[10];
    const float* b2     = (const float*)d_in[11];
    float* out = (float*)d_out;

    char* ws = (char*)d_ws;
    const size_t MB = 1ull << 20;
    __bf16* t_bf  = (__bf16*)(ws);                       // 1 MB
    float*  base1 = (float*)(ws + 1 * MB);               // 8 MB
    float*  base3 = (float*)(ws + 9 * MB);               // 8 MB
    float*  u1    = (float*)(ws + 17 * MB);              // 256 KB
    float*  u3    = (float*)(ws + 17 * MB + 262144);     // 256 KB
    __bf16* act   = (__bf16*)(ws + 18 * MB);             // 8 MB (2 slots)
    float*  bout  = (float*)(ws + 26 * MB);              // 4 MB (2 slots)
    float*  vbuf  = (float*)(ws + 30 * MB);              // 64 KB
    int*    tok_i = (int*)(ws + 30 * MB + 65536);        // 4 KB
    float*  tok_w = (float*)(ws + 30 * MB + 65536 + 4096);
    int*    perm  = (int*)(ws + 30 * MB + 65536 + 8192); // 4 KB
    int*    eoff  = (int*)(ws + 30 * MB + 65536 + 12288);
    int*    ecnt  = (int*)(ws + 30 * MB + 65536 + 16384);

    // 1) RMSNorm + router
    norm_router_kernel<<<T_TOK, 256, 0, stream>>>(x, norm_w, gate_w, t_bf, tok_i, tok_w);

    // 2) group (token,slot) pairs by expert; zero atomic accumulator
    sort_kernel<<<1, 256, 0, stream>>>(tok_i, eoff, ecnt, perm);
    zero_kernel<<<(2 * T_TOK * RANK + 255) / 256, 256, 0, stream>>>(vbuf, 2 * T_TOK * RANK);

    // 3) LoRA up intermediates u1/u3 (WMMA)
    lora_u_kernel<<<dim3(T_TOK / 16, N_EXP, 2), 32, 0, stream>>>(t_bf, a1, a3, u1, u3);

    // 4) base1/base3 = t @ w{1,3}^T  (WMMA, 512x4096x1024, 64x256 block tile)
    wmma_gemm_abt<256, 2, 4><<<dim3(DIM_F / 256, T_TOK / 64), 256, 0, stream>>>(
        t_bf, w1, base1, T_TOK, DIM_F, DIM_D);
    wmma_gemm_abt<256, 2, 4><<<dim3(DIM_F / 256, T_TOK / 64), 256, 0, stream>>>(
        t_bf, w3, base3, T_TOK, DIM_F, DIM_D);

    // 5) SwiGLU with per-token-expert LoRA deltas (expert-grouped)
    act_sorted_kernel<<<dim3(N_EXP, DIM_F / 64), 256, 0, stream>>>(
        base1, base3, u1, u3, b1, b3, eoff, ecnt, perm, act);

    // 6) down base: bout[s] = act[s] @ w2^T  (WMMA, 512x1024x4096, 64x128 tile)
    wmma_gemm_abt<128, 2, 2><<<dim3(DIM_D / 128, T_TOK / 64), 256, 0, stream>>>(
        act, w2, bout, T_TOK, DIM_D, DIM_F);
    wmma_gemm_abt<128, 2, 2><<<dim3(DIM_D / 128, T_TOK / 64), 256, 0, stream>>>(
        act + (size_t)T_TOK * DIM_F, w2, bout + (size_t)T_TOK * DIM_D,
        T_TOK, DIM_D, DIM_F);

    // 7) down LoRA intermediates v (expert-grouped, atomic accumulate)
    v_sorted_kernel<<<dim3(N_EXP, DIM_F / 256), 256, 0, stream>>>(
        act, a2, eoff, ecnt, perm, vbuf);

    // 8) combine with router weights
    combine_kernel<<<T_TOK, 256, 0, stream>>>(bout, vbuf, b2, tok_i, tok_w, out);
}
